// GraphConvolution_84997402788523
// MI455X (gfx1250) — matrix operand order repaired and checked by reading, not compiled, not run on hardware
//
#include <hip/hip_runtime.h>

// ---------------------------------------------------------------------------
// GraphConvolution on MI455X (gfx1250):
//   support = context @ Wctx^T            (bf16 WMMA, f32 accum, NT loads of A)
//   agg     = COO scatter-add of support  (global_atomic_add_f32, L2-resident)
//   out     = relu(agg) @ Wout^T + b      (bf16 WMMA, fused relu/bias, NT store)
//
// Tile: BM=64, BN=256, BK=32; 256 threads = 8 wave32; each wave computes a
// 16x128 strip (8 x v_wmma_f32_16x16x32_bf16 per K-step). With BN=256,
// GEMM1 streams the 409.6MB context from HBM exactly once (grid.y == 1).
// ---------------------------------------------------------------------------

typedef __attribute__((ext_vector_type(16))) __bf16 v16bf;
typedef __attribute__((ext_vector_type(8)))  __bf16 v8bf;
typedef __attribute__((ext_vector_type(8)))  float  v8f;

#define TILE_M 64
#define TILE_N 256
#define TILE_K 32
#define LDS_STRIDE 40  // bf16 elems/row: 80B stride keeps 16B-aligned K segments

template <bool RELU_A, bool ADD_BIAS, bool NT_A, bool NT_STORE>
__global__ __launch_bounds__(256) void gemm_bt_wmma_kernel(
    const float* __restrict__ A,     // [M,K] row-major
    const float* __restrict__ W,     // [N,K] row-major  (B = W^T)
    const float* __restrict__ bias,  // [N] (when ADD_BIAS)
    float* __restrict__ C,           // [M,N] row-major
    int M, int N, int K) {
  __shared__ __align__(16) __bf16 ldsA[TILE_M][LDS_STRIDE];
  __shared__ __align__(16) __bf16 ldsB[TILE_N][LDS_STRIDE];

  const int tid  = threadIdx.x;
  const int lane = tid & 31;
  const int wid  = tid >> 5;  // 0..7
  const int wm   = wid & 3;   // 16-row slice of the 64-row tile
  const int wn   = wid >> 2;  // 128-col half of the 256-col tile

  const int m0 = blockIdx.x * TILE_M;
  const int n0 = blockIdx.y * TILE_N;

  // A staging: 4 threads per row, 8 f32 each (64 x 32)
  const int lrow = tid >> 2;       // 0..63
  const int lcol = (tid & 3) * 8;  // 0,8,16,24

  const bool hi  = lane >= 16;
  const int  l16 = lane & 15;

  const int  gmA   = m0 + lrow;
  const bool rowOK = gmA < M;
  const float keep = rowOK ? 1.0f : 0.0f;               // branch-free M tail
  const float* __restrict__ srcA =
      A + (size_t)(rowOK ? gmA : 0) * K + lcol;         // clamped address
  const float* __restrict__ srcB = W + (size_t)(n0 + tid) * K;

  v8f acc[8];
#pragma unroll
  for (int s = 0; s < 8; ++s) acc[s] = (v8f){};

  for (int k0 = 0; k0 < K; k0 += TILE_K) {
    // ---- stage A tile (f32 -> bf16, optional relu, branch-free pad) ----
#pragma unroll
    for (int i = 0; i < 8; ++i) {
      float x = NT_A ? __builtin_nontemporal_load(srcA + k0 + i) : srcA[k0 + i];
      x *= keep;
      if (RELU_A) x = fmaxf(x, 0.0f);
      ldsA[lrow][lcol + i] = (__bf16)x;
    }
    // ---- stage B tile: one W row per thread, 32 contiguous f32 ----
#pragma unroll
    for (int i = 0; i < 32; ++i) ldsB[tid][i] = (__bf16)srcB[k0 + i];
    __syncthreads();

    // speculative prefetch of next A K-tile (global_prefetch_b8)
    if (k0 + TILE_K < K) __builtin_prefetch(srcA + k0 + TILE_K, 0, 0);

    // ---- A fragment (16x32 bf16): lanes 0-15 K{0..7,16..23},
    //      lanes 16-31 K{8..15,24..31}, row M = lane%16 ----
    const __bf16* ar = &ldsA[wm * 16 + l16][0];
    v8bf a0 = *(const v8bf*)(ar + (hi ? 8 : 0));
    v8bf a1 = *(const v8bf*)(ar + (hi ? 24 : 16));
    v16bf afrag;
#pragma unroll
    for (int i = 0; i < 8; ++i) {
      afrag[i]     = a0[i];
      afrag[i + 8] = a1[i];
    }

    // ---- 8 B fragments + 8 WMMAs: wave strip is 16 x 128 ----
#pragma unroll
    for (int s = 0; s < 8; ++s) {
      const __bf16* br = &ldsB[wn * 128 + s * 16 + l16][0] + (hi ? 8 : 0);
      v8bf b0 = *(const v8bf*)(br);
      v8bf b1 = *(const v8bf*)(br + 16);
      v16bf bfrag;
#pragma unroll
      for (int i = 0; i < 8; ++i) {
        bfrag[i]     = b0[i];
        bfrag[i + 8] = b1[i];
      }
      acc[s] = __builtin_amdgcn_wmma_f32_16x16x32_bf16(
          false, afrag, false, bfrag, (short)0, acc[s], false, false);
    }
    __syncthreads();
  }

  // ---- epilogue: VGPR r -> M = r + (hi?8:0); N = lane%16 ----
  const int rbase = m0 + wm * 16 + (hi ? 8 : 0);
#pragma unroll
  for (int s = 0; s < 8; ++s) {
    const int col = n0 + wn * 128 + s * 16 + l16;
    if (col >= N) continue;
    const float bv = ADD_BIAS ? bias[col] : 0.0f;
#pragma unroll
    for (int r = 0; r < 8; ++r) {
      const int grow = rbase + r;
      if (grow < M) {
        const float v = acc[s][r] + bv;
        float* dst = &C[(size_t)grow * N + col];
        if (NT_STORE)
          __builtin_nontemporal_store(v, dst);
        else
          *dst = v;
      }
    }
  }
}

__global__ void zero_kernel(float* __restrict__ p, size_t n) {
  size_t i = (size_t)blockIdx.x * blockDim.x + threadIdx.x;
  const size_t stride = (size_t)gridDim.x * blockDim.x;
  for (; i < n; i += stride) p[i] = 0.0f;
}

// one wave32 per edge; lanes stripe the 256-wide hid vector; support+agg
// (102 MB) are L2-resident, so gathers and f32 atomic scatters resolve in L2
__global__ __launch_bounds__(256) void spmm_scatter_kernel(
    const long long* __restrict__ rows, const long long* __restrict__ cols,
    const float* __restrict__ vals, const float* __restrict__ support,
    float* __restrict__ agg, int E, int H) {
  const int e = blockIdx.x * 8 + (threadIdx.x >> 5);
  if (e >= E) return;
  const int lane = threadIdx.x & 31;
  const size_t r = (size_t)rows[e];
  const size_t c = (size_t)cols[e];
  const float v = vals[e];
  const float4* __restrict__ src = (const float4*)(support + c * H);
  float* __restrict__ dst = agg + r * H;
  for (int i = lane; i < (H >> 2); i += 32) {
    const float4 s = src[i];
    atomicAdd(dst + i * 4 + 0, s.x * v);
    atomicAdd(dst + i * 4 + 1, s.y * v);
    atomicAdd(dst + i * 4 + 2, s.z * v);
    atomicAdd(dst + i * 4 + 3, s.w * v);
  }
}

extern "C" void kernel_launch(void* const* d_in, const int* in_sizes, int n_in,
                              void* d_out, int out_size, void* d_ws, size_t ws_size,
                              hipStream_t stream) {
  const float*     context = (const float*)d_in[0];
  const long long* adj_row = (const long long*)d_in[1];
  const long long* adj_col = (const long long*)d_in[2];
  const float*     adj_val = (const float*)d_in[3];
  const float*     ctx_w   = (const float*)d_in[4];  // [HID, VOCAB]
  const float*     out_w   = (const float*)d_in[5];  // [VOCAB, HID]
  const float*     out_b   = (const float*)d_in[6];  // [VOCAB]
  float* out = (float*)d_out;

  const int VOCAB  = in_sizes[6];
  const int HID    = in_sizes[5] / VOCAB;
  const int NNODES = in_sizes[0] / VOCAB;
  const int E      = in_sizes[1];

  float* support = (float*)d_ws;                    // [NNODES, HID]
  float* agg     = support + (size_t)NNODES * HID;  // [NNODES, HID]

  const dim3 blk(256);

  // 1) support = context @ ctx_w^T   (M=NNODES, N=HID, K=VOCAB)
  //    grid.y == 1 -> context streamed from HBM exactly once (NT loads);
  //    support kept in L2 (regular stores) for the SpMM stage.
  dim3 g1((NNODES + TILE_M - 1) / TILE_M, (HID + TILE_N - 1) / TILE_N);
  gemm_bt_wmma_kernel<false, false, true, false><<<g1, blk, 0, stream>>>(
      context, ctx_w, nullptr, support, NNODES, HID, VOCAB);

  // 2) agg = 0
  zero_kernel<<<2048, 256, 0, stream>>>(agg, (size_t)NNODES * HID);

  // 3) agg[row] += val * support[col]
  spmm_scatter_kernel<<<(E + 7) / 8, blk, 0, stream>>>(
      adj_row, adj_col, adj_val, support, agg, E, HID);

  // 4) out = relu(agg) @ out_w^T + out_b   (M=NNODES, N=VOCAB, K=HID)
  //    agg + out_w are L2-resident (regular loads); 409.6MB output is
  //    write-once -> non-temporal stores.
  dim3 g4((NNODES + TILE_M - 1) / TILE_M, (VOCAB + TILE_N - 1) / TILE_N);
  gemm_bt_wmma_kernel<true, true, false, true><<<g4, blk, 0, stream>>>(
      agg, out_w, out_b, out, NNODES, VOCAB, HID);
}
